// QuantumPolicyNetwork_4818953306144
// MI455X (gfx1250) — matrix operand order repaired and checked by reading, not compiled, not run on hardware
//
#include <hip/hip_runtime.h>
#include <hip/hip_bf16.h>
#include <math.h>

typedef __bf16 bf16;
typedef __attribute__((ext_vector_type(16))) __bf16 v16bf;
typedef __attribute__((ext_vector_type(8)))  __bf16 v8bf;
typedef __attribute__((ext_vector_type(4)))  __bf16 v4bf;
typedef __attribute__((ext_vector_type(8)))  float  v8f;
typedef __attribute__((ext_vector_type(4)))  float  v4f;

#define B_TOT   8192
#define S_DIM   64
#define A_DIM   26
#define D_DIM   4096
#define H1_DIM  128
#define H2_DIM  64
#define HA_DIM  128
#define AP_DIM  52
#define M_TILE  16
#define N_TILES (B_TOT / M_TILE)   // 512 workgroups
#define CHUNK   512
#define NCHUNK  (D_DIM / CHUNK)    // 8
#define THREADS 256

// ---------------- LDS layout (dynamic shared) ----------------
#define SM_F_LOGITS 0
#define SM_F_STATE  (SM_F_LOGITS + M_TILE * D_DIM)
#define SM_F_H1     (SM_F_STATE + M_TILE * S_DIM)
#define SM_F_HA     (SM_F_H1 + M_TILE * H1_DIM)
#define SM_F_SCALE  (SM_F_HA + M_TILE * HA_DIM)
#define SM_F_KROW   (SM_F_SCALE + 16)
#define SM_F_INVN   (SM_F_KROW + 16)
#define SM_F_DIV    (SM_F_INVN + 16)
#define SM_F_ENTW   (SM_F_DIV + 16)
#define SM_F_TOTAL  (SM_F_ENTW + 8)            // 70728 floats (16B-aligned end)
#define SM_BF_H2    0
#define SM_BF_MEAS  (SM_BF_H2 + M_TILE * H2_DIM)
#define SM_BF_TOTAL (SM_BF_MEAS + M_TILE * CHUNK)
#define SMEM_BYTES  (SM_F_TOTAL * 4 + SM_BF_TOTAL * 2)   // ~294 KB <= 320 KB/WGP

// ---------------- wave32 reductions ----------------
__device__ __forceinline__ float wave_sum(float v) {
#pragma unroll
  for (int off = 16; off > 0; off >>= 1) v += __shfl_xor(v, off, 32);
  return v;
}
__device__ __forceinline__ float wave_max(float v) {
#pragma unroll
  for (int off = 16; off > 0; off >>= 1) v = fmaxf(v, __shfl_xor(v, off, 32));
  return v;
}

// ---------------- WMMA fragment loaders (ISA 7.12.2), vectorized ----------------
// A 16x32 bf16: lane<16 -> row=lane, elems 0..7 = K(k0+0..7),  8..15 = K(k0+16..23)
//               lane>=16 -> row=lane-16, elems 0..7 = K(k0+8..15), 8..15 = K(k0+24..31)
// -> two contiguous 8-element (16B) runs per lane.
__device__ __forceinline__ v16bf load_a_frag(const bf16* base, int lda, int k0, int lane) {
  const int row = lane & 15;
  const int hlf = lane >> 4;
  const bf16* p = base + row * lda + k0 + 8 * hlf;
  v8bf lo = *(const v8bf*)(p);
  v8bf hi = *(const v8bf*)(p + 16);
  return __builtin_shufflevector(lo, hi, 0, 1, 2, 3, 4, 5, 6, 7,
                                         8, 9, 10, 11, 12, 13, 14, 15);
}
// B 32x16 bf16 from K-contiguous transposed weights WT[N][K]:
// col = lane&15, elems 0..7 -> K = k0 + 8*(lane>>4) + e, 8..15 -> K = k0 + 16 + 8*(lane>>4) + (e-8).
__device__ __forceinline__ v16bf load_bT_frag(const bf16* baseT, int ldk, int k0, int n0, int lane) {
  const int col = lane & 15;
  const int hlf = lane >> 4;
  const bf16* p = baseT + (size_t)(n0 + col) * ldk + k0 + 8 * hlf;
  v8bf lo = *(const v8bf*)(p);
  v8bf hi = *(const v8bf*)(p + 16);
  return __builtin_shufflevector(lo, hi, 0, 1, 2, 3, 4, 5, 6, 7,
                                         8, 9, 10, 11, 12, 13, 14, 15);
}

// ---------------- kernel 1: scalars (gate magnitude, coherence) ----------------
__global__ void scalar_kernel(const float* __restrict__ rot, const float* __restrict__ ec,
                              float* __restrict__ wsF, float* __restrict__ out) {
  if (threadIdx.x == 0 && blockIdx.x == 0) {
    float gmag = 1.0f;
    for (int i = 0; i < 36; ++i) {  // L*NQ = 3*12
      float h0 = 0.5f * rot[i * 3 + 0];
      float h1 = 0.5f * rot[i * 3 + 1];
      float h2 = 0.5f * rot[i * 3 + 2];
      float cr = __cosf(h0) * __cosf(h1) * __cosf(h2);
      float si = __sinf(h0) * __sinf(h1) * __sinf(h2);
      gmag *= sqrtf(cr * cr + si * si);   // |e^{i sum(ent)}| == 1
    }
    wsF[0] = gmag;
    wsF[1] = 1.0f / (1.0f + __expf(-ec[0]));  // coh = sigmoid
    out[2 * B_TOT * A_DIM + 0] = ec[0];       // returned unchanged by reference
  }
}

// ---------------- kernel 2: bf16 + transpose weight prep (one-time, L2-resident) ----
// WqT:  [N=4096][K=64]   from Wq  [64][4096]
// Wa1T: [N=128][K=4096]  from Wa1 [4096][128]
__global__ void prep_kernel(const float* __restrict__ Wq, const float* __restrict__ Wa1,
                            bf16* __restrict__ wsWqT, bf16* __restrict__ wsWa1T) {
  const int stride = gridDim.x * blockDim.x;
  const int i0 = blockIdx.x * blockDim.x + threadIdx.x;
  for (int o = i0; o < D_DIM * H2_DIM; o += stride) {
    int n = o >> 6, k = o & 63;                   // o = n*64 + k
    wsWqT[o] = (bf16)Wq[(size_t)k * D_DIM + n];
  }
  for (int o = i0; o < HA_DIM * D_DIM; o += stride) {
    int n = o >> 12, k = o & 4095;                // o = n*4096 + k
    wsWa1T[o] = (bf16)Wa1[(size_t)k * HA_DIM + n];
  }
}

// ---------------- kernel 3: fused per-16-row pipeline ----------------
__global__ void __launch_bounds__(THREADS, 1)
fused_kernel(const float* __restrict__ state, const float* __restrict__ phases,
             const float* __restrict__ nre, const float* __restrict__ nim,
             const float* __restrict__ We1, const float* __restrict__ be1,
             const float* __restrict__ We2, const float* __restrict__ be2,
             const bf16* __restrict__ WqT_bf, const float* __restrict__ bq,
             const bf16* __restrict__ Wa1T_bf, const float* __restrict__ ba1,
             const float* __restrict__ Wa2, const float* __restrict__ ba2,
             const float* __restrict__ wsF, float* __restrict__ partials,
             float* __restrict__ out) {
  extern __shared__ char smem_raw[];
  float* smF = (float*)smem_raw;
  float* sLogits = smF + SM_F_LOGITS;
  float* sState  = smF + SM_F_STATE;
  float* sH1     = smF + SM_F_H1;
  float* sHa     = smF + SM_F_HA;
  float* sScale  = smF + SM_F_SCALE;
  float* sKRow   = smF + SM_F_KROW;
  float* sInvN   = smF + SM_F_INVN;
  float* sDivRow = smF + SM_F_DIV;
  float* sEntW   = smF + SM_F_ENTW;
  bf16* smB   = (bf16*)(smF + SM_F_TOTAL);
  bf16* sH2   = smB + SM_BF_H2;
  bf16* sMeas = smB + SM_BF_MEAS;

  const int tid  = threadIdx.x;
  const int lane = tid & 31;
  const int wave = tid >> 5;
  const int wg   = blockIdx.x;
  const int rowBase = wg * M_TILE;
  const float gmag = wsF[0];
  const float coh  = wsF[1];

  // ---- load state tile ----
  for (int i = tid; i < M_TILE * S_DIM; i += THREADS)
    sState[i] = state[(size_t)rowBase * S_DIM + i];
  __syncthreads();

  // ---- encoder layer 1 (VALU, tiny) ----
  for (int o = tid; o < M_TILE * H1_DIM; o += THREADS) {
    int r = o >> 7, c = o & 127;
    float acc = be1[c];
#pragma unroll 8
    for (int j = 0; j < S_DIM; ++j) acc += sState[r * S_DIM + j] * We1[j * H1_DIM + c];
    sH1[o] = fmaxf(acc, 0.0f);
  }
  __syncthreads();

  // ---- encoder layer 2 -> bf16 for WMMA A ----
  for (int o = tid; o < M_TILE * H2_DIM; o += THREADS) {
    int r = o >> 6, c = o & 63;
    float acc = be2[c];
#pragma unroll 8
    for (int j = 0; j < H1_DIM; ++j) acc += sH1[r * H1_DIM + j] * We2[j * H2_DIM + c];
    sH2[o] = (bf16)fmaxf(acc, 0.0f);
  }
  __syncthreads();

  // ---- GEMM3: logits = h2(16x64) @ W_q(64x4096) + b_q  [WMMA bf16] ----
  {
    v16bf a0 = load_a_frag(sH2, H2_DIM, 0, lane);
    v16bf a1 = load_a_frag(sH2, H2_DIM, 32, lane);
    const int col = lane & 15;
    const int hlf = lane >> 4;
    for (int t = 0; t < D_DIM / (16 * 8); ++t) {   // 32 tiles per wave
      int n0 = wave * (D_DIM / 8) + t * 16;
      float bias = bq[n0 + col];
      v8f acc;
#pragma unroll
      for (int r = 0; r < 8; ++r) acc[r] = bias;
      v16bf b0 = load_bT_frag(WqT_bf, H2_DIM, 0, n0, lane);
      acc = __builtin_amdgcn_wmma_f32_16x16x32_bf16(false, a0, false, b0, (short)0, acc, false, false);
      v16bf b1 = load_bT_frag(WqT_bf, H2_DIM, 32, n0, lane);
      acc = __builtin_amdgcn_wmma_f32_16x16x32_bf16(false, a1, false, b1, (short)0, acc, false, false);
#pragma unroll
      for (int r = 0; r < 8; ++r) {
        int m = r + (hlf ? 8 : 0);
        sLogits[m * D_DIM + n0 + col] = acc[r];
      }
    }
  }
  __syncthreads();

  // ---- softmax (in place, keep unnormalized e) + diversity + pass-A norm sums ----
  // each wave owns rows 2*wave and 2*wave+1
  for (int rr = 0; rr < 2; ++rr) {
    const int r = wave * 2 + rr;
    float* row = sLogits + r * D_DIM;
    const size_t gbase = (size_t)(rowBase + r) * D_DIM;

    float mx = -1e30f;
    for (int d = lane; d < D_DIM; d += 32) mx = fmaxf(mx, row[d]);
    mx = wave_max(mx);

    float s = 0.0f, s4 = 0.0f;
    for (int d = lane; d < D_DIM; d += 32) {
      float e = __expf(row[d] - mx);
      row[d] = e;
      s += e;
      float e2 = e * e;
      s4 += e2 * e2;
    }
    s  = wave_sum(s);
    s4 = wave_sum(s4);
    const float scale = coh / s;          // u_d = scale * e_d = coh * amps_d

    // pass A: S1 = sum u^2, S2 = sum |noise|^2, S3 = sum u.(noise proj)
    float S1 = 0.0f, S2 = 0.0f, S3 = 0.0f;
    const v4f* ph4 = (const v4f*)(phases + gbase);
    const v4f* xr4 = (const v4f*)(nre + gbase);
    const v4f* xi4 = (const v4f*)(nim + gbase);
    const v4f* e4p = (const v4f*)row;
    for (int i = lane; i < D_DIM / 4; i += 32) {
      v4f ev = e4p[i];
      v4f pv = ph4[i];
      v4f rv = xr4[i];
      v4f iv = xi4[i];
#pragma unroll
      for (int j = 0; j < 4; ++j) {
        float u = scale * ev[j];
        S1 += u * u;
        S2 += rv[j] * rv[j] + iv[j] * iv[j];
        S3 += u * (__cosf(pv[j]) * rv[j] + __sinf(pv[j]) * iv[j]);
      }
    }
    S1 = wave_sum(S1); S2 = wave_sum(S2); S3 = wave_sum(S3);
    const float kf = (1.0f - coh) / fmaxf(sqrtf(S2), 1e-12f);
    const float V  = S1 + kf * kf * S2 + 2.0f * kf * S3;   // ||v||^2
    const float invn = 1.0f / fmaxf(sqrtf(V), 1e-12f);
    if (lane == 0) {
      sScale[r] = scale;
      sKRow[r]  = kf;
      sInvN[r]  = invn;
      float ss = s * s;
      sDivRow[r] = (ss * ss) / s4;        // 1 / sum(amps^4)
    }
  }
  __syncthreads();

  // ---- pass B: chunked measured tiles + GEMM4 (WMMA, persistent accumulators) ----
  v8f acc4;
#pragma unroll
  for (int r = 0; r < 8; ++r) acc4[r] = 0.0f;
  const float biasA1 = ba1[wave * 16 + (lane & 15)];
  float entAcc = 0.0f;

  for (int c = 0; c < NCHUNK; ++c) {
    for (int rr = 0; rr < 2; ++rr) {
      const int r = wave * 2 + rr;
      const float scale = sScale[r];
      const float kf    = sKRow[r];
      const float invn  = sInvN[r];
      const float in2   = invn * invn;
      const size_t gb   = (size_t)(rowBase + r) * D_DIM + (size_t)c * CHUNK;
      const v4f* e4p = (const v4f*)(sLogits + r * D_DIM + c * CHUNK);
      const v4f* ph4 = (const v4f*)(phases + gb);
      const v4f* xr4 = (const v4f*)(nre + gb);
      const v4f* xi4 = (const v4f*)(nim + gb);
      v4bf* mst = (v4bf*)(sMeas + r * CHUNK);
      // prefetch next chunk of this row's streaming data into cache
      if (c + 1 < NCHUNK) {
        __builtin_prefetch(phases + gb + CHUNK + lane * 16, 0, 1);
        __builtin_prefetch(nre + gb + CHUNK + lane * 16, 0, 1);
        __builtin_prefetch(nim + gb + CHUNK + lane * 16, 0, 1);
      }
      for (int i = lane; i < CHUNK / 4; i += 32) {
        v4f ev = e4p[i];
        v4f pv = ph4[i];
        v4f rv = xr4[i];
        v4f iv = xi4[i];
        v4bf mv;
#pragma unroll
        for (int j = 0; j < 4; ++j) {
          float u  = scale * ev[j];
          float re = u * __cosf(pv[j]) + kf * rv[j];
          float im = u * __sinf(pv[j]) + kf * iv[j];
          float m2 = re * re + im * im;
          float p  = m2 * in2 + 1e-8f;
          entAcc += p * __logf(p);
          mv[j] = (bf16)(gmag * sqrtf(m2) * invn);
        }
        mst[i] = mv;
      }
    }
    __syncthreads();
    // GEMM4 chunk: each wave owns 16 output cols, K = 512 per chunk
    const int n0w = wave * 16;
#pragma unroll 4
    for (int kk = 0; kk < CHUNK / 32; ++kk) {
      v16bf a = load_a_frag(sMeas, CHUNK, kk * 32, lane);
      v16bf b = load_bT_frag(Wa1T_bf, D_DIM, c * CHUNK + kk * 32, n0w, lane);
      acc4 = __builtin_amdgcn_wmma_f32_16x16x32_bf16(false, a, false, b, (short)0, acc4, false, false);
    }
    __syncthreads();
  }

  // ---- ha = relu(acc + b_a1), entropy wave totals ----
  {
    const int col = lane & 15;
    const int hlf = lane >> 4;
#pragma unroll
    for (int r = 0; r < 8; ++r) {
      int m = r + (hlf ? 8 : 0);
      sHa[m * HA_DIM + wave * 16 + col] = fmaxf(acc4[r] + biasA1, 0.0f);
    }
  }
  entAcc = wave_sum(entAcc);
  if (lane == 0) sEntW[wave] = entAcc;
  __syncthreads();

  // ---- GEMM5 head (VALU, tiny) + outputs ----
  for (int o = tid; o < M_TILE * AP_DIM; o += THREADS) {
    int r = o / AP_DIM, cc = o % AP_DIM;
    float acc = ba2[cc];
#pragma unroll 8
    for (int j = 0; j < HA_DIM; ++j) acc += sHa[r * HA_DIM + j] * Wa2[j * AP_DIM + cc];
    int grow = rowBase + r;
    if (cc < A_DIM) out[(size_t)grow * A_DIM + cc] = acc;
    else            out[(size_t)B_TOT * A_DIM + (size_t)grow * A_DIM + (cc - A_DIM)] = acc;
  }

  if (tid == 0) {
    float ent = 0.0f;
    for (int w = 0; w < 8; ++w) ent += sEntW[w];
    float dv = 0.0f;
    for (int r = 0; r < M_TILE; ++r) dv += sDivRow[r];
    partials[wg * 2 + 0] = -ent;   // sum over tile rows of -sum(p log p)
    partials[wg * 2 + 1] = dv;
  }
}

// ---------------- kernel 4: deterministic ordered final reduction ----------------
__global__ void final_kernel(const float* __restrict__ partials, float* __restrict__ out) {
  if (threadIdx.x == 0 && blockIdx.x == 0) {
    float ent = 0.0f, dv = 0.0f;
    for (int i = 0; i < N_TILES; ++i) { ent += partials[i * 2]; dv += partials[i * 2 + 1]; }
    out[2 * B_TOT * A_DIM + 1] = ent / (float)B_TOT;
    out[2 * B_TOT * A_DIM + 2] = dv / (float)B_TOT;
  }
}

extern "C" void kernel_launch(void* const* d_in, const int* in_sizes, int n_in,
                              void* d_out, int out_size, void* d_ws, size_t ws_size,
                              hipStream_t stream) {
  const float* state  = (const float*)d_in[0];
  const float* phases = (const float*)d_in[1];
  const float* nre    = (const float*)d_in[2];
  const float* nim    = (const float*)d_in[3];
  const float* We1    = (const float*)d_in[4];
  const float* be1    = (const float*)d_in[5];
  const float* We2    = (const float*)d_in[6];
  const float* be2    = (const float*)d_in[7];
  const float* Wq     = (const float*)d_in[8];
  const float* bq     = (const float*)d_in[9];
  const float* Wa1    = (const float*)d_in[10];
  const float* ba1    = (const float*)d_in[11];
  const float* Wa2    = (const float*)d_in[12];
  const float* ba2    = (const float*)d_in[13];
  const float* rot    = (const float*)d_in[14];
  // d_in[15] entangling_params: only phase -> |gate| magnitude unaffected
  const float* ec     = (const float*)d_in[16];
  float* out = (float*)d_out;

  float* wsF      = (float*)d_ws;           // [0]=gate mag, [1]=coh
  float* partials = wsF + 16;               // 512*2 floats
  bf16*  wsWqT    = (bf16*)((char*)d_ws + 4 * (16 + 2 * N_TILES));
  bf16*  wsWa1T   = wsWqT + D_DIM * H2_DIM;

  (void)hipFuncSetAttribute((const void*)fused_kernel,
                            hipFuncAttributeMaxDynamicSharedMemorySize, SMEM_BYTES);

  scalar_kernel<<<1, 32, 0, stream>>>(rot, ec, wsF, out);
  prep_kernel<<<512, THREADS, 0, stream>>>(Wq, Wa1, wsWqT, wsWa1T);
  fused_kernel<<<N_TILES, THREADS, SMEM_BYTES, stream>>>(
      state, phases, nre, nim, We1, be1, We2, be2,
      wsWqT, bq, wsWa1T, ba1, Wa2, ba2, wsF, partials, out);
  final_kernel<<<1, 32, 0, stream>>>(partials, out);
}
